// MonarchTransformer_35424890257841
// MI455X (gfx1250) — compile-verified, matrix operand order checked by reading, not run on hardware
//
#include <hip/hip_runtime.h>
#include <cstdint>

typedef __attribute__((ext_vector_type(16))) __bf16 bf16x16;
typedef __attribute__((ext_vector_type(8)))  __bf16 bf16x8;
typedef __attribute__((ext_vector_type(8)))  float  f32x8;
typedef __attribute__((ext_vector_type(4)))  float  f32x4;

#define TOKENS  8192
#define HEADS   16
#define QKDIM   64
#define VDIM    960
#define RFEAT   128

#define AC 0.1f
#define BC 0.7745966692414834f     /* sqrt(1 - 4*0.1) */
#define CC (-1.0f)                  /* -(S+1)/2, S=1 */
#define DC 2.821109907456e-4f       /* ((1-0.4)^0.25)^64 = 0.6^16 */
#define LNEPS 1e-5f

__device__ __forceinline__ void block_reduce2(float* red1, float* red2,
                                              float a, float b, int t) {
  __syncthreads();                 // protect reuse of red arrays
  red1[t] = a; red2[t] = b;
  __syncthreads();
  for (int s = 128; s > 0; s >>= 1) {
    if (t < s) { red1[t] += red1[t + s]; red2[t] += red2[t + s]; }
    __syncthreads();
  }
}

// ---------------------------------------------------------------------------
// Kernel A: per-token front end. 256 threads (8 waves) per token.
//  z = einsum(nij,jik->njk) ; o = einsum(njk,kjl->nkl) ; leaky relu
//  LN(Q over 1024), LN(K over 64), LN(V over 960) ; RoPE ; /sqrt(64)
//  wQ via WMMA bf16 ; f1 -> bf16, f2 -> f32, V -> f32
// ---------------------------------------------------------------------------
__global__ __launch_bounds__(256) void ka_frontend(
    const float* __restrict__ data, const float* __restrict__ locations,
    const float* __restrict__ L,    const float* __restrict__ R,
    const float* __restrict__ q_scale, const float* __restrict__ q_bias,
    const float* __restrict__ k_scale, const float* __restrict__ k_bias,
    const float* __restrict__ v_scale, const float* __restrict__ v_bias,
    const float* __restrict__ W,
    __bf16* __restrict__ f1buf, float* __restrict__ f2buf,
    float* __restrict__ vbuf)
{
  const int n = blockIdx.x;
  const int t = threadIdx.x;

  __shared__ float xs[2048];     // x row; later reused as Qs (scaled, rotated)
  __shared__ float zs[2048];
  __shared__ float qkv[2048];
  __shared__ float red1[256], red2[256];
  __shared__ float karr[64], ksarr[64];
  __shared__ float cosv[32], sinv[32];
  __shared__ float sq2[16];
  __shared__ float stats[1];

  // 1. load x row (reshape (64,32): x[i][j] = data[n, i*32+j])
  for (int i = t; i < 2048; i += 256) xs[i] = data[(size_t)n * 2048 + i];
  __syncthreads();

  // 2. z[j][k] = sum_i x[i][j] * L[j,i,k]
  for (int o8 = 0; o8 < 8; ++o8) {
    int out = t + o8 * 256;
    int j = out >> 6, k = out & 63;
    const float* Lp = L + ((size_t)j * 64) * 64 + k;
    float acc = 0.f;
    for (int i = 0; i < 64; ++i) acc = fmaf(xs[i * 32 + j], Lp[i * 64], acc);
    zs[out] = acc;                       // zs[j*64+k]
  }
  __syncthreads();

  // 3. o[k][l] = sum_j z[j][k] * R[k,j,l], leaky relu -> qkv[k*32+l]
  for (int o8 = 0; o8 < 8; ++o8) {
    int out = t + o8 * 256;
    int k = out >> 5, l = out & 31;
    const float* Rp = R + ((size_t)k * 32) * 32 + l;
    float acc = 0.f;
    for (int j = 0; j < 32; ++j) acc = fmaf(zs[j * 64 + k], Rp[j * 32], acc);
    qkv[out] = acc > 0.f ? acc : 0.1f * acc;
  }
  __syncthreads();

  // 4. layernorm statistics
  float s1 = 0.f, s2 = 0.f;
  for (int i = t; i < 1024; i += 256) { float v = qkv[i]; s1 += v; s2 += v * v; }
  block_reduce2(red1, red2, s1, s2, t);
  float muQ = red1[0] * (1.f / 1024.f);
  float rsQ = rsqrtf(red2[0] * (1.f / 1024.f) - muQ * muQ + LNEPS);

  s1 = s2 = 0.f;
  if (t < 64) { float v = qkv[1024 + t]; s1 = v; s2 = v * v; }
  block_reduce2(red1, red2, s1, s2, t);
  float muK = red1[0] * (1.f / 64.f);
  float rsK = rsqrtf(red2[0] * (1.f / 64.f) - muK * muK + LNEPS);

  s1 = s2 = 0.f;
  for (int i = 1088 + t; i < 2048; i += 256) { float v = qkv[i]; s1 += v; s2 += v * v; }
  block_reduce2(red1, red2, s1, s2, t);
  float muV = red1[0] * (1.f / 960.f);
  float rsV = rsqrtf(red2[0] * (1.f / 960.f) - muV * muV + LNEPS);

  // 5. normalized V -> global, K -> karr, Q -> xs (reuse)
  for (int i = t; i < 960; i += 256) {
    float v = (qkv[1088 + i] - muV) * rsV * v_scale[i] + v_bias[i];
    vbuf[(size_t)n * 960 + i] = v;
  }
  if (t < 64) karr[t] = (qkv[1024 + t] - muK) * rsK * k_scale[t] + k_bias[t];
  for (int i = t; i < 1024; i += 256)
    xs[i] = (qkv[i] - muQ) * rsQ * q_scale[i] + q_bias[i];
  if (t < 32) {
    // inv_freq = exp(-t * ln(10000)/32)
    float ang = locations[n] * __expf(-(float)t * 0.28782313662425572f);
    cosv[t] = cosf(ang);
    sinv[t] = sinf(ang);
  }
  __syncthreads();

  // 6. RoPE + 1/sqrt(QKDIM) scaling (in place; disjoint pairs per thread)
  if (t < 32) {
    float c = cosv[t], s = sinv[t];
    float k1 = karr[t], k2 = karr[t + 32];
    ksarr[t]      = (c * k1 + s * k2) * 0.125f;   // sign = +1
    ksarr[t + 32] = (c * k2 - s * k1) * 0.125f;
  }
  for (int i = t; i < 512; i += 256) {
    int h = i >> 5, d = i & 31;
    float c = cosv[d], s = sinv[d];
    float q1 = xs[h * 64 + d], q2 = xs[h * 64 + d + 32];
    xs[h * 64 + d]      = (c * q1 - s * q2) * 0.125f;  // sign = -1
    xs[h * 64 + d + 32] = (c * q2 + s * q1) * 0.125f;
  }
  __syncthreads();

  // 7. per-head |Qs|^2 and |Ks|^2
  if (t < 16) {
    float a = 0.f;
    for (int d = 0; d < 64; ++d) { float q = xs[t * 64 + d]; a += q * q; }
    sq2[t] = a;
  }
  if (t == 16) {
    float a = 0.f;
    for (int d = 0; d < 64; ++d) { float k = ksarr[d]; a += k * k; }
    stats[0] = a;
  }
  __syncthreads();
  const float sumK2 = stats[0];

  // 8. wK + f2 (VALU; 128 outputs)
  if (t < 128) {
    const float* Wp = W + (size_t)t * 64;
    float wk = 0.f;
    for (int d = 0; d < 64; ++d) wk = fmaf(ksarr[d], Wp[d], wk);
    f2buf[(size_t)n * 128 + t] = DC * __expf(AC + BC * wk + CC * sumK2);
  }

  // 9. wQ via WMMA bf16: 16x64 (Qs) x 64x128 (W^T) -> 16x128 ; each wave one
  //    16-column tile, K split as 2 x 32.
  {
    const int lane = t & 31, wave = t >> 5;
    const int r0 = wave * 16;
    const int m = lane & 15, half = lane >> 4, base = half * 8;
    f32x8 acc;
#pragma unroll
    for (int e = 0; e < 8; ++e) acc[e] = 0.f;
#pragma unroll
    for (int kt = 0; kt < 2; ++kt) {
      const int k0 = kt * 32 + base;
      bf16x16 a, b;
#pragma unroll
      for (int e = 0; e < 8; ++e) {
        a[e]     = (__bf16)xs[m * 64 + k0 + e];        // K = k0..k0+7
        a[8 + e] = (__bf16)xs[m * 64 + k0 + 16 + e];   // K = k0+16..k0+23
      }
      const float* Wp = W + (size_t)(r0 + m) * 64 + k0;
#pragma unroll
      for (int e = 0; e < 8; ++e) {
        b[e]     = (__bf16)Wp[e];
        b[8 + e] = (__bf16)Wp[16 + e];
      }
      acc = __builtin_amdgcn_wmma_f32_16x16x32_bf16(
          false, a, false, b, (short)0, acc, false, false);
    }
    // C layout: col = r0 + lane%16 ; row(head) = e + 8*(lane/16)
#pragma unroll
    for (int e = 0; e < 8; ++e) {
      int h = e + 8 * half;
      int r = r0 + m;
      float f1v = DC * __expf(AC + BC * acc[e] + CC * sq2[h]);
      f1buf[((size_t)n * 16 + h) * 128 + r] = (__bf16)f1v;
    }
  }
}

// ---------------------------------------------------------------------------
// Kernel B1: deterministic split-N partial kv. chunk c covers 256 tokens.
// pk[c][r][v] = sum_{n in chunk} f2[n][r] * V[n][v]
// grid (32 chunks, 8 r-tiles, 15 v-tiles), 256 threads.
// ---------------------------------------------------------------------------
__global__ __launch_bounds__(256) void kb1_partial(
    const float* __restrict__ f2buf, const float* __restrict__ vbuf,
    float* __restrict__ pk)
{
  const int c = blockIdx.x, rt = blockIdx.y, vt = blockIdx.z;
  const int t = threadIdx.x;
  const int r = rt * 16 + (t & 15);
  const int v0 = vt * 64 + (t >> 4) * 4;
  float a0 = 0.f, a1 = 0.f, a2 = 0.f, a3 = 0.f;
  const int n0 = c * 256;
  for (int nn = 0; nn < 256; ++nn) {
    const int n = n0 + nn;
    const float f = f2buf[(size_t)n * 128 + r];
    const f32x4 vv = *(const f32x4*)(vbuf + (size_t)n * 960 + v0);
    a0 = fmaf(f, vv.x, a0);
    a1 = fmaf(f, vv.y, a1);
    a2 = fmaf(f, vv.z, a2);
    a3 = fmaf(f, vv.w, a3);
  }
  float* o = pk + ((size_t)c * 128 + r) * 960 + v0;
  o[0] = a0; o[1] = a1; o[2] = a2; o[3] = a3;
}

// ---------------------------------------------------------------------------
// Kernel B2: reduce 32 chunk partials (fixed order) and emit kv directly in
// WMMA B-fragment order (bf16):
//   kvw[((vt*4 + kt)*32 + lane)*16 + e]
//   lane -> col v = vt*16 + lane%16 ; base = (lane>=16)?8:0
//   e<8  -> K row r = kt*32 + base + e
//   e>=8 -> K row r = kt*32 + base + 16 + (e-8)
// 480 blocks x 256 = 122880 threads = 4*60*32*16.
// ---------------------------------------------------------------------------
__global__ __launch_bounds__(256) void kb2_reduce_swizzle(
    const float* __restrict__ pk, __bf16* __restrict__ kvw)
{
  const int tid = blockIdx.x * 256 + threadIdx.x;
  const int e = tid & 15;
  const int lane = (tid >> 4) & 31;
  const int kt = (tid >> 9) & 3;
  const int vt = tid >> 11;
  const int base = (lane >= 16) ? 8 : 0;
  const int r = kt * 32 + base + ((e < 8) ? e : (8 + e));
  const int v = vt * 16 + (lane & 15);
  float s = 0.f;
  for (int c = 0; c < 32; ++c) s += pk[((size_t)c * 128 + r) * 960 + v];
  kvw[tid] = (__bf16)s;
}

// ---------------------------------------------------------------------------
// Kernel C: O = f1 (131072 x 128, bf16) x kv (128 x 960, bf16) -> f32.
// 8 waves/block; wave owns 16 rows x 64 cols (4 accumulators), K = 4 x 32.
// grid (1024 row groups, 15 col groups). NT stores for the 503 MB output.
// ---------------------------------------------------------------------------
__global__ __launch_bounds__(256) void kc_gemm(
    const __bf16* __restrict__ f1buf, const __bf16* __restrict__ kvw,
    float* __restrict__ out)
{
  const int t = threadIdx.x, lane = t & 31, wave = t >> 5;
  const int rowTile = blockIdx.x * 8 + wave;   // 0..8191 (rows = n*16+h)
  const int colBlk = blockIdx.y;               // 0..14
  const int m = lane & 15, half = lane >> 4, base = half * 8;

  f32x8 acc[4];
#pragma unroll
  for (int ct = 0; ct < 4; ++ct)
#pragma unroll
    for (int e = 0; e < 8; ++e) acc[ct][e] = 0.f;

  const __bf16* ap = f1buf + ((size_t)rowTile * 16 + m) * 128;
#pragma unroll
  for (int kt = 0; kt < 4; ++kt) {
    bf16x16 a;
    const bf16x8 lo = *(const bf16x8*)(ap + kt * 32 + base);
    const bf16x8 hi = *(const bf16x8*)(ap + kt * 32 + base + 16);
#pragma unroll
    for (int e = 0; e < 8; ++e) { a[e] = lo[e]; a[8 + e] = hi[e]; }
#pragma unroll
    for (int ct = 0; ct < 4; ++ct) {
      const int vt = colBlk * 4 + ct;
      const bf16x16 b =
          *(const bf16x16*)(kvw + ((size_t)(vt * 4 + kt) * 32 + lane) * 16);
      acc[ct] = __builtin_amdgcn_wmma_f32_16x16x32_bf16(
          false, a, false, b, (short)0, acc[ct], false, false);
    }
  }

  const int colBase = colBlk * 64;
#pragma unroll
  for (int ct = 0; ct < 4; ++ct) {
#pragma unroll
    for (int e = 0; e < 8; ++e) {
      const size_t row = (size_t)rowTile * 16 + e + 8 * half;
      const int col = colBase + ct * 16 + m;
      __builtin_nontemporal_store(acc[ct][e], out + row * 960 + col);
    }
  }
}

// ---------------------------------------------------------------------------
// Workspace layout (bytes):
//   f1buf  bf16  [0,          33554432)   8192*16*128
//   f2buf  f32   [33554432,   37748736)   8192*128
//   vbuf   f32   [37748736,   69206016)   8192*960
//   pk     f32   [69206016,   84934656)   32*128*960
//   kvw    bf16  [84934656,   85180416)   4*60*32*16
// total ~85.2 MB. Every workspace element is fully written each call before
// being read (no memset / no atomics), so replays are deterministic.
// ---------------------------------------------------------------------------
extern "C" void kernel_launch(void* const* d_in, const int* in_sizes, int n_in,
                              void* d_out, int out_size, void* d_ws, size_t ws_size,
                              hipStream_t stream)
{
  (void)in_sizes; (void)n_in; (void)out_size; (void)ws_size;
  const float* data      = (const float*)d_in[0];
  const float* locations = (const float*)d_in[1];
  const float* L         = (const float*)d_in[2];
  const float* R         = (const float*)d_in[3];
  const float* q_scale   = (const float*)d_in[4];
  const float* q_bias    = (const float*)d_in[5];
  const float* k_scale   = (const float*)d_in[6];
  const float* k_bias    = (const float*)d_in[7];
  const float* v_scale   = (const float*)d_in[8];
  const float* v_bias    = (const float*)d_in[9];
  const float* W         = (const float*)d_in[10];

  char* ws = (char*)d_ws;
  __bf16* f1buf = (__bf16*)(ws);
  float*  f2buf = (float*)(ws + 33554432);
  float*  vbuf  = (float*)(ws + 37748736);
  float*  pk    = (float*)(ws + 69206016);
  __bf16* kvw   = (__bf16*)(ws + 84934656);

  ka_frontend<<<TOKENS, 256, 0, stream>>>(
      data, locations, L, R, q_scale, q_bias, k_scale, k_bias,
      v_scale, v_bias, W, f1buf, f2buf, vbuf);

  kb1_partial<<<dim3(32, 8, 15), 256, 0, stream>>>(f2buf, vbuf, pk);

  kb2_reduce_swizzle<<<480, 256, 0, stream>>>(pk, kvw);

  kc_gemm<<<dim3(1024, 15), 256, 0, stream>>>(f1buf, kvw, (float*)d_out);
}
